// Level_Cat_3_8_16767552324153
// MI455X (gfx1250) — matrix operand order repaired and checked by reading, not compiled, not run on hardware
//
#include <hip/hip_runtime.h>
#include <hip/hip_bf16.h>

typedef __attribute__((ext_vector_type(16))) _Float16 v16h;
typedef __attribute__((ext_vector_type(8)))  _Float16 v8h;
typedef __attribute__((ext_vector_type(8)))  float    v8f;
typedef __attribute__((ext_vector_type(4)))  unsigned int v4u;

#define HW     4096   // H*W
#define NP     1024   // pooled positions
#define CIN    64
#define CINT   32
#define BATCH  16
#define WAVES  8      // waves per workgroup in the attention kernel

#if defined(__AMDGCN__) && __has_builtin(__builtin_amdgcn_global_load_async_to_lds_b128)
#define HAS_ASYNC_LDS 1
// builtin signature (from hipcc diagnostic): (int4 AS1*, <lds ptr>, imm, imm)
typedef int v4i __attribute__((vector_size(16)));
typedef __attribute__((address_space(1))) v4i gv4i;
typedef __attribute__((address_space(3))) v4i lv4i;
#else
#define HAS_ASYNC_LDS 0
#endif

// ---------------------------------------------------------------------------
// K1: K = maxpool(phi(x)) + maxpool(hc(level_feature))  -> [B][1024][32] f16
//     Vt = maxpool(g(x)) transposed                     -> [B][32][1024] f16
// thread = (pooled position, feature); weights staged in LDS (padded strides
// to avoid 32-way bank conflicts on the f-indexed reads).
// ---------------------------------------------------------------------------
__global__ __launch_bounds__(256)
void prep_kv_kernel(const float* __restrict__ x, const float* __restrict__ lf,
                    const float* __restrict__ hc_w, const float* __restrict__ hc_b,
                    const float* __restrict__ g_w,  const float* __restrict__ g_b,
                    const float* __restrict__ phi_w,const float* __restrict__ phi_b,
                    _Float16* __restrict__ Kmat, _Float16* __restrict__ Vt) {
  __shared__ float sphi[32 * 65];
  __shared__ float sg[32 * 65];
  __shared__ float shc[32 * 33];
  for (int i = threadIdx.x; i < 32 * 64; i += 256) {
    int f = i >> 6, c = i & 63;
    sphi[f * 65 + c] = phi_w[i];
    sg[f * 65 + c]   = g_w[i];
  }
  for (int i = threadIdx.x; i < 32 * 32; i += 256) {
    int f = i >> 5, c = i & 31;
    shc[f * 33 + c] = hc_w[i];
  }
  __syncthreads();

  int t  = blockIdx.x * 256 + threadIdx.x;
  int f  = t & 31;
  int p  = t >> 5;            // b*1024 + pp
  int b  = p >> 10;
  int pp = p & 1023;
  int h2 = pp >> 5, w2 = pp & 31;
  int nbase = (h2 * 2) * 64 + (w2 * 2);
  const float* xb = x  + (size_t)b * CIN * HW;
  const float* lb = lf + (size_t)b * 32  * HW;

  float phim = -1e30f, gm = -1e30f, hm = -1e30f;
  const int offs[4] = {0, 1, 64, 65};
#pragma unroll
  for (int px = 0; px < 4; ++px) {
    int n = nbase + offs[px];
    float sp = 0.f, sv = 0.f;
#pragma unroll 8
    for (int c = 0; c < 64; ++c) {
      float xv = xb[c * HW + n];
      sp += xv * sphi[f * 65 + c];
      sv += xv * sg[f * 65 + c];
    }
    float sh = 0.f;
#pragma unroll 8
    for (int c = 0; c < 32; ++c) sh += lb[c * HW + n] * shc[f * 33 + c];
    phim = fmaxf(phim, sp);
    gm   = fmaxf(gm, sv);
    hm   = fmaxf(hm, sh);
  }
  // additive dual-source key: th@ph + th@lf == th@(ph+lf)
  Kmat[((size_t)b * NP + pp) * 32 + f] = (_Float16)(phim + phi_b[f] + hm + hc_b[f]);
  Vt[((size_t)b * 32 + f) * NP + pp]   = (_Float16)(gm + g_b[f]);
}

// ---------------------------------------------------------------------------
// K2: Q = theta(x) -> [B][4096][32] f16 (position-major, feature contiguous)
// ---------------------------------------------------------------------------
__global__ __launch_bounds__(256)
void theta_kernel(const float* __restrict__ x,
                  const float* __restrict__ th_w, const float* __restrict__ th_b,
                  _Float16* __restrict__ Q) {
  __shared__ float sw[32 * 65];
  for (int i = threadIdx.x; i < 32 * 64; i += 256) {
    int f = i >> 6, c = i & 63;
    sw[f * 65 + c] = th_w[i];
  }
  __syncthreads();
  int t = blockIdx.x * 256 + threadIdx.x;
  int f = t & 31;
  int p = t >> 5;             // b*4096 + n
  int b = p >> 12;
  int n = p & 4095;
  const float* xb = x + (size_t)b * CIN * HW;
  float s = 0.f;
#pragma unroll 8
  for (int c = 0; c < 64; ++c) s += xb[c * HW + n] * sw[f * 65 + c];
  Q[((size_t)b * HW + n) * 32 + f] = (_Float16)(s + th_b[f]);
}

// W_w [64][32] f32 -> f16 (already in B^T layout: out-channel rows, contiguous K)
__global__ void convert_w_kernel(const float* __restrict__ W_w, _Float16* __restrict__ Wh) {
  int i = blockIdx.x * 256 + threadIdx.x;
  if (i < 64 * 32) Wh[i] = (_Float16)W_w[i];
}

// ---------------------------------------------------------------------------
// K3: fused flash attention + output projection + BN + residual.
// 8 waves per WG; each wave owns 16 query rows (128 rows/WG). K[1024][32] and
// Vt[32][1024] resident in LDS (128 KB of the 320 KB WGP pool), filled with
// GLOBAL_LOAD_ASYNC_TO_LDS_B128 (ASYNCcnt) when available. Online softmax
// without max-subtraction (scores bounded << 88). All GEMMs are
// v_wmma_f32_16x16x32_f16 (K dim == inter == 32).
// ---------------------------------------------------------------------------
__global__ __launch_bounds__(WAVES * 32)
void attention_kernel(const _Float16* __restrict__ Q,
                      const _Float16* __restrict__ Kmat,
                      const _Float16* __restrict__ Vt,
                      const _Float16* __restrict__ Wh,
                      const float* __restrict__ x,
                      const float* __restrict__ W_b,
                      const float* __restrict__ bn_gamma, const float* __restrict__ bn_beta,
                      const float* __restrict__ bn_mean,  const float* __restrict__ bn_var,
                      float* __restrict__ out) {
  extern __shared__ _Float16 smem[];
  _Float16* Kl  = smem;            // 1024*32 halves (64 KB)
  _Float16* Vtl = smem + 32768;    // 32*1024 halves (64 KB)
  const int wave = threadIdx.x >> 5;
  const int lane = threadIdx.x & 31;
  _Float16* Pst = smem + 65536 + wave * 512;   // per-wave 16x32 staging (1 KB)

  const int b      = blockIdx.x >> 5;          // 32 row-blocks per batch
  const int rowblk = blockIdx.x & 31;

  // cooperative fill of K, V into LDS (b128 granules)
  {
#if HAS_ASYNC_LDS
    gv4i* gKa = (gv4i*)(void*)(Kmat + (size_t)b * NP * 32);
    gv4i* gVa = (gv4i*)(void*)(Vt   + (size_t)b * 32 * NP);
    lv4i* lKa = (lv4i*)(void*)Kl;
    lv4i* lVa = (lv4i*)(void*)Vtl;
    for (int i = threadIdx.x; i < 4096; i += WAVES * 32) {
      __builtin_amdgcn_global_load_async_to_lds_b128(gKa + i, lKa + i, 0, 0);
      __builtin_amdgcn_global_load_async_to_lds_b128(gVa + i, lVa + i, 0, 0);
    }
    asm volatile("s_wait_asynccnt 0x0" ::: "memory");
#else
    const v4u* gK = (const v4u*)(Kmat + (size_t)b * NP * 32);
    const v4u* gV = (const v4u*)(Vt   + (size_t)b * 32 * NP);
    v4u* lK = (v4u*)Kl;
    v4u* lV = (v4u*)Vtl;
    for (int i = threadIdx.x; i < 4096; i += WAVES * 32) { lK[i] = gK[i]; lV[i] = gV[i]; }
#endif
  }
  __syncthreads();

  const int m0   = rowblk * (WAVES * 16) + wave * 16;  // first query row of wave
  const int mrow = lane & 15;                // tile row (A) / tile col (B,C,D)
  const int hi   = lane >> 4;

  // Q A-tile: lanes<16 hold K0-15, lanes>=16 hold K16-31
  v16h aQ;
  {
    const _Float16* qrow = Q + ((size_t)b * HW + m0 + mrow) * 32;
    union { v16h v; v8h h[2]; } u;
    u.h[0] = *(const v8h*)(qrow + (hi ? 16 : 0));
    u.h[1] = *(const v8h*)(qrow + (hi ? 24 : 8));
    aQ = u.v;
  }

  v8f acc0 = {};                 // Y features 0..15
  v8f acc1 = {};                 // Y features 16..31
  float psum[8] = {0, 0, 0, 0, 0, 0, 0, 0};

  for (int n0 = 0; n0 < NP; n0 += 32) {
    // --- S = Q K^T for 32 key positions; P = exp(S) staged to LDS ---
#pragma unroll
    for (int t = 0; t < 2; ++t) {
      const int nt = n0 + t * 16;
      v16h bK = *(const v16h*)(Kl + (nt + mrow) * 32 + hi * 16);
      v8f s = {};
      s = __builtin_amdgcn_wmma_f32_16x16x32_f16(false, aQ, false, bK,
                                                 (short)0, s, false, false);
#pragma unroll
      for (int r = 0; r < 8; ++r) {
        float e = __expf(s[r]);
        psum[r] += e;                         // running row-sum (per-lane partial)
        Pst[(r + hi * 8) * 32 + t * 16 + mrow] = (_Float16)e;
      }
    }
    asm volatile("s_wait_dscnt 0x0" ::: "memory");   // stage stores -> A-tile loads

    v16h aP;
    {
      const _Float16* prow = Pst + mrow * 32;
      union { v16h v; v8h h[2]; } u;
      u.h[0] = *(const v8h*)(prow + (hi ? 16 : 0));
      u.h[1] = *(const v8h*)(prow + (hi ? 24 : 8));
      aP = u.v;
    }
    // --- acc += P V for this 32-key chunk ---
    v16h bV0 = *(const v16h*)(Vtl + (0  + mrow) * NP + n0 + hi * 16);
    v16h bV1 = *(const v16h*)(Vtl + (16 + mrow) * NP + n0 + hi * 16);
    acc0 = __builtin_amdgcn_wmma_f32_16x16x32_f16(false, aP, false, bV0,
                                                  (short)0, acc0, false, false);
    acc1 = __builtin_amdgcn_wmma_f32_16x16x32_f16(false, aP, false, bV1,
                                                  (short)0, acc1, false, false);
  }

  // row sums: reduce over the 16 lanes of each half (rows r / r+8 match the
  // C/D layout exactly, so the reciprocal applies directly to acc VGPRs)
#pragma unroll
  for (int r = 0; r < 8; ++r) {
    float v = psum[r];
    v += __shfl_xor(v, 1, 16);
    v += __shfl_xor(v, 2, 16);
    v += __shfl_xor(v, 4, 16);
    v += __shfl_xor(v, 8, 16);
    psum[r] = 1.0f / v;
  }

  // normalized Y -> staging (row-major 16x32), reload as A-tile
#pragma unroll
  for (int r = 0; r < 8; ++r) {
    const int m = r + hi * 8;
    Pst[m * 32 + mrow]      = (_Float16)(acc0[r] * psum[r]);
    Pst[m * 32 + 16 + mrow] = (_Float16)(acc1[r] * psum[r]);
  }
  asm volatile("s_wait_dscnt 0x0" ::: "memory");
  v16h aY;
  {
    const _Float16* yrow = Pst + mrow * 32;
    union { v16h v; v8h h[2]; } u;
    u.h[0] = *(const v8h*)(yrow + (hi ? 16 : 0));
    u.h[1] = *(const v8h*)(yrow + (hi ? 24 : 8));
    aY = u.v;
  }

  // fused output projection (32 -> 64) + BatchNorm + residual
#pragma unroll
  for (int ot = 0; ot < 4; ++ot) {
    const int o = ot * 16 + mrow;
    v16h bW = *(const v16h*)(Wh + o * 32 + hi * 16);
    v8f od = {};
    od = __builtin_amdgcn_wmma_f32_16x16x32_f16(false, aY, false, bW,
                                                (short)0, od, false, false);
    const float scale = bn_gamma[o] * rsqrtf(bn_var[o] + 1e-5f);
    const float cadd  = (W_b[o] - bn_mean[o]) * scale + bn_beta[o];
    const float* xr   = x   + ((size_t)b * 64 + o) * HW + m0;
    float*       orow = out + ((size_t)b * 64 + o) * HW + m0;
#pragma unroll
    for (int r = 0; r < 8; ++r) {
      const int m = r + hi * 8;
      const float res = __builtin_nontemporal_load(&xr[m]);
      __builtin_nontemporal_store(od[r] * scale + cadd + res, &orow[m]);
    }
  }
}

// ---------------------------------------------------------------------------
extern "C" void kernel_launch(void* const* d_in, const int* in_sizes, int n_in,
                              void* d_out, int out_size, void* d_ws, size_t ws_size,
                              hipStream_t stream) {
  (void)in_sizes; (void)n_in; (void)out_size; (void)ws_size;
  const float* x     = (const float*)d_in[0];
  const float* lf    = (const float*)d_in[1];
  const float* hc_w  = (const float*)d_in[2];
  const float* hc_b  = (const float*)d_in[3];
  const float* g_w   = (const float*)d_in[4];
  const float* g_b   = (const float*)d_in[5];
  const float* th_w  = (const float*)d_in[6];
  const float* th_b  = (const float*)d_in[7];
  const float* phi_w = (const float*)d_in[8];
  const float* phi_b = (const float*)d_in[9];
  const float* W_w   = (const float*)d_in[10];
  const float* W_b   = (const float*)d_in[11];
  const float* gam   = (const float*)d_in[12];
  const float* bet   = (const float*)d_in[13];
  const float* mea   = (const float*)d_in[14];
  const float* var   = (const float*)d_in[15];
  float* out = (float*)d_out;

  char* ws = (char*)d_ws;
  _Float16* Q    = (_Float16*)(ws);                              // 4 MB
  _Float16* Kmat = (_Float16*)(ws + (size_t)4 * 1024 * 1024);    // 1 MB
  _Float16* Vt   = (_Float16*)(ws + (size_t)5 * 1024 * 1024);    // 1 MB
  _Float16* Wh   = (_Float16*)(ws + (size_t)6 * 1024 * 1024);    // 4 KB

  convert_w_kernel<<<8, 256, 0, stream>>>(W_w, Wh);
  prep_kv_kernel<<<2048, 256, 0, stream>>>(x, lf, hc_w, hc_b, g_w, g_b,
                                           phi_w, phi_b, Kmat, Vt);
  theta_kernel<<<8192, 256, 0, stream>>>(x, th_w, th_b, Q);

  // 16 batches * 32 row-blocks; 8 waves (256 threads); ~136 KB dynamic LDS
  const size_t smemBytes = (size_t)(65536 + WAVES * 512) * sizeof(_Float16);
  attention_kernel<<<dim3(BATCH * 32), dim3(WAVES * 32), smemBytes, stream>>>(
      Q, Kmat, Vt, Wh, x, W_b, gam, bet, mea, var, out);
}
